// L_color_weighted_79594333929489
// MI455X (gfx1250) — compile-verified, matrix operand order checked by reading, not run on hardware
//
#include <hip/hip_runtime.h>
#include <math.h>

// SLIC-superpixel-weighted colorfulness, 16 images of 3x512x512 f32.
// Multi-kernel pipeline (all on `stream`, graph-capture safe):
//   init -> 11 x assign (512 blocks: 32 blocks/image) interleaved with
//   10 x update -> finalize.
// Assignment distances via V_WMMA_F32_16X16X32_F16 (16 pixels x 64 centers,
// 4 column tiles); argmin as branch-free packed-u64 min reduction over
// ds_bpermute shuffles; segment sums via ds_add_f32 then global_atomic_add_f32.

#define K_SEG 50
#define ITERS 10
#define IMG_H 512
#define IMG_W 512
#define NPIX (IMG_H * IMG_W)
#define TILES (NPIX / 16)        // 16384 16-pixel tiles per image
#define BPI 32                   // blocks per image in assign pass
#define TPB (TILES / BPI)        // tiles per block = 512
#define CSTRIDE (64 * 5)         // padded centers per image (64 cols x 5 dims)
#define ASTRIDE (K_SEG * 6)      // per image: 50 x {sum0..sum4, count}

typedef __attribute__((ext_vector_type(16))) _Float16 v16h;
typedef __attribute__((ext_vector_type(8)))  float    v8f;

__device__ __forceinline__ float slic_ratio() {
    return 10.0f / sqrtf((float)NPIX / (float)K_SEG);   // compactness / S
}

// ---------------- init: grid-sampled centers, zeroed accumulators ----------------
__global__ __launch_bounds__(256)
void slic_init_kernel(const float* __restrict__ x, float* __restrict__ gcent,
                      float* __restrict__ gacc)
{
    const int b = blockIdx.x;
    const float* img = x + (size_t)b * 3u * NPIX;
    float* cent = gcent + b * CSTRIDE;
    float* acc  = gacc  + b * ASTRIDE;
    const float ratio = slic_ratio();

    for (int i = threadIdx.x; i < CSTRIDE; i += blockDim.x) cent[i] = 0.0f;
    for (int i = threadIdx.x; i < ASTRIDE; i += blockDim.x) acc[i]  = 0.0f;
    __syncthreads();
    if (threadIdx.x < K_SEG) {
        int gi = threadIdx.x >> 3, gj = threadIdx.x & 7;       // 8x8 grid, first 50
        int cy = 32 + 64 * gi, cx = 32 + 64 * gj;
        int p = cy * IMG_W + cx;
        cent[threadIdx.x * 5 + 0] = img[0 * NPIX + p];
        cent[threadIdx.x * 5 + 1] = img[1 * NPIX + p];
        cent[threadIdx.x * 5 + 2] = img[2 * NPIX + p];
        cent[threadIdx.x * 5 + 3] = (float)cy * ratio;
        cent[threadIdx.x * 5 + 4] = (float)cx * ratio;
    }
}

// ---------------- assign: labels + segment sums for one pass ----------------
__global__ __launch_bounds__(256)
void slic_assign_kernel(const float* __restrict__ x,
                        const float* __restrict__ gcent,
                        float* __restrict__ gacc)
{
    const int b     = blockIdx.x / BPI;
    const int chunk = blockIdx.x % BPI;
    const float* img  = x + (size_t)b * 3u * NPIX;
    const float* cent = gcent + b * CSTRIDE;
    float* acc        = gacc  + b * ASTRIDE;
    const float ratio = slic_ratio();

    __shared__ float lacc[ASTRIDE];          // 50 x {5 sums, count}

    const int tid    = threadIdx.x;
    const int lane   = tid & 31;             // wave32
    const int wave   = tid >> 5;
    const int nwaves = blockDim.x >> 5;
    const int grp    = lane >> 4;
    const int l      = lane & 15;

    for (int i = tid; i < ASTRIDE; i += blockDim.x) lacc[i] = 0.0f;

    // ---- B fragments (centers -> f16) and c^2 per column, per lane ----
    // 16-bit B 32x16 layout: lanes 0-15 carry K=0-7 (elements 0..7) for N=l.
    v16h bfrag[4];
    float c2v[4];
    #pragma unroll
    for (int t = 0; t < 4; ++t) {
        const int cidx = t * 16 + l;
        float s = 0.0f;
        v16h bb = {};
        #pragma unroll
        for (int d = 0; d < 5; ++d) {
            float c = cent[cidx * 5 + d];
            s += c * c;
            if (grp == 0) bb[d] = (_Float16)c;
        }
        bfrag[t] = bb;
        c2v[t] = (cidx < K_SEG) ? s : 3.0e38f;   // padded columns never win
    }
    __syncthreads();

    const int tile_lo = chunk * TPB;
    const int tile_hi = tile_lo + TPB;

    for (int tile = tile_lo + wave; tile < tile_hi; tile += nwaves) {
        const int base = tile * 16;

        if (tile + nwaves < tile_hi)          // speculative global_prefetch_b8
            __builtin_prefetch(img + (base + nwaves * 16 + l), 0, 0);

        // ---- A fragment: lane l (grp0) holds pixel base+l, dims at K=0..4 ----
        v16h a = {};
        float f0 = 0.f, f1 = 0.f, f2 = 0.f, f3 = 0.f, f4 = 0.f;
        {
            const int p  = base + l;
            const int py = p >> 9;
            const int px = p & 511;
            if (grp == 0) {
                f0 = img[0 * NPIX + p];
                f1 = img[1 * NPIX + p];
                f2 = img[2 * NPIX + p];
                f3 = (float)py * ratio;
                f4 = (float)px * ratio;
                a[0] = (_Float16)f0; a[1] = (_Float16)f1; a[2] = (_Float16)f2;
                a[3] = (_Float16)f3; a[4] = (_Float16)f4;
            }
        }

        // ---- packed (score,index) argmin keys; D[v] is pixel v+8*grp, col l ----
        unsigned long long pk[8];
        #pragma unroll
        for (int v = 0; v < 8; ++v) pk[v] = 0xFFFFFFFFFFFFFFFFull;

        #pragma unroll
        for (int t = 0; t < 4; ++t) {
            v8f dacc = {};
            dacc = __builtin_amdgcn_wmma_f32_16x16x32_f16(
                false, a, false, bfrag[t], (short)0, dacc, false, false);
            const unsigned cidx = (unsigned)(t * 16 + l);
            #pragma unroll
            for (int v = 0; v < 8; ++v) {
                float s = fmaf(-2.0f, dacc[v], c2v[t]);
                unsigned bits = __float_as_uint(s);
                // order-preserving uint transform (handles negative scores)
                unsigned u = bits ^ ((unsigned)((int)bits >> 31) | 0x80000000u);
                unsigned long long cand = ((unsigned long long)u << 32) | cidx;
                pk[v] = (cand < pk[v]) ? cand : pk[v];      // branch-free
            }
        }

        // ---- branch-free min reduction within each 16-lane half ----
        #pragma unroll
        for (int m = 1; m <= 8; m <<= 1) {
            #pragma unroll
            for (int v = 0; v < 8; ++v) {
                unsigned long long o = __shfl_xor(pk[v], m, 32);
                pk[v] = (o < pk[v]) ? o : pk[v];
            }
        }
        // exchange halves: grp0 reg v -> pixel v, grp1 reg v -> pixel 8+v
        int own[8], oth[8];
        #pragma unroll
        for (int v = 0; v < 8; ++v) {
            unsigned long long o = __shfl_xor(pk[v], 16, 32);
            own[v] = (int)(pk[v] & 63u);
            oth[v] = (int)(o & 63u);
        }

        if (grp == 0) {
            // pick label of pixel m = l (registers only, no scratch)
            const int  want = l & 7;
            const bool hi   = (l >= 8);
            int lab = 0;
            #pragma unroll
            for (int v = 0; v < 8; ++v) {
                int cand = hi ? oth[v] : own[v];
                lab = (want == v) ? cand : lab;
            }
            float* s = &lacc[lab * 6];
            atomicAdd(s + 0, f0);
            atomicAdd(s + 1, f1);
            atomicAdd(s + 2, f2);
            atomicAdd(s + 3, f3);
            atomicAdd(s + 4, f4);
            atomicAdd(s + 5, 1.0f);
        }
    }

    __syncthreads();
    // flush block partials to global accumulators
    for (int i = tid; i < ASTRIDE; i += blockDim.x)
        atomicAdd(&acc[i], lacc[i]);
}

// ---------------- update: new centers = sums / max(cnt,1); re-zero ----------------
__global__ __launch_bounds__(64)
void slic_update_kernel(float* __restrict__ gcent, float* __restrict__ gacc)
{
    const int b = blockIdx.x;
    float* cent = gcent + b * CSTRIDE;
    float* acc  = gacc  + b * ASTRIDE;
    const int t = threadIdx.x;
    if (t < K_SEG) {
        float inv = 1.0f / fmaxf(acc[t * 6 + 5], 1.0f);
        #pragma unroll
        for (int d = 0; d < 5; ++d) cent[t * 5 + d] = acc[t * 6 + d] * inv;
        #pragma unroll
        for (int d = 0; d < 6; ++d) acc[t * 6 + d] = 0.0f;
    }
}

// ---------------- finalize: weighted mean_rgb -> colorfulness scalar ----------------
__global__ __launch_bounds__(32)
void slic_final_kernel(const float* __restrict__ gacc, float* __restrict__ out)
{
    const int b = blockIdx.x;
    const float* acc = gacc + b * ASTRIDE;
    if (threadIdx.x == 0) {
        float mr = 0.f, mg = 0.f, mb = 0.f;
        for (int j = 0; j < K_SEG; ++j) {
            float inv = 1.0f / fmaxf(acc[j * 6 + 5], 1.0f);
            mr += acc[j * 6 + 0] * inv;
            mg += acc[j * 6 + 1] * inv;
            mb += acc[j * 6 + 2] * inv;
        }
        const float invN = 1.0f / (float)NPIX;
        mr *= invN; mg *= invN; mb *= invN;
        float Drg = (mr - mg) * (mr - mg);
        float Drb = (mr - mb) * (mr - mb);
        float Dgb = (mb - mg) * (mb - mg);
        out[b] = sqrtf(Drg * Drg + Drb * Drb + Dgb * Dgb);
    }
}

extern "C" void kernel_launch(void* const* d_in, const int* in_sizes, int n_in,
                              void* d_out, int out_size, void* d_ws, size_t ws_size,
                              hipStream_t stream) {
    (void)n_in; (void)ws_size; (void)out_size;
    const float* x = (const float*)d_in[0];
    float* out = (float*)d_out;
    const int batch = in_sizes[0] / (3 * NPIX);     // 16

    float* gcent = (float*)d_ws;                    // batch * 320 floats
    float* gacc  = gcent + (size_t)batch * CSTRIDE; // batch * 300 floats

    slic_init_kernel<<<batch, 256, 0, stream>>>(x, gcent, gacc);
    for (int p = 0; p < ITERS; ++p) {
        slic_assign_kernel<<<batch * BPI, 256, 0, stream>>>(x, gcent, gacc);
        slic_update_kernel<<<batch, 64, 0, stream>>>(gcent, gacc);
    }
    slic_assign_kernel<<<batch * BPI, 256, 0, stream>>>(x, gcent, gacc);
    slic_final_kernel<<<batch, 32, 0, stream>>>(gacc, out);
}
